// NottinghamModel_26903675142834
// MI455X (gfx1250) — matrix operand order repaired and checked by reading, not compiled, not run on hardware
//
#include <hip/hip_runtime.h>
#include <cstdint>
#include <cstddef>

#define T_STEPS 256
#define BATCH   128
#define DIM     128
#define HID     1024
#define MEL     48

#define BN 64          // N tile per block
#define KC 32          // K per chunk (one WMMA depth)
#define ROWE 40        // padded LDS row length in bf16 (80B, kills bank conflicts)

typedef __bf16 bf16_t;
typedef __attribute__((ext_vector_type(16))) __bf16 v16bf;
typedef __attribute__((ext_vector_type(8)))  __bf16 v8bf;
typedef __attribute__((ext_vector_type(8)))  float  v8f;
typedef int v4i __attribute__((vector_size(16)));   // GCC-vector int4 (builtin pointee type)

__device__ __forceinline__ bf16_t f2bf(float f) {
  unsigned u = __builtin_bit_cast(unsigned, f);
  unsigned r = u + 0x7FFFu + ((u >> 16) & 1u);   // round-to-nearest-even
  unsigned short h = (unsigned short)(r >> 16);
  return __builtin_bit_cast(bf16_t, h);
}

__device__ __forceinline__ float sigm(float x) { return 1.0f / (1.0f + __expf(-x)); }

// --- CDNA5 async VMEM->LDS copy (ASYNCcnt-tracked), with portable fallback ---
__device__ __forceinline__ void async_copy16(const bf16_t* g, bf16_t* l) {
#if __has_builtin(__builtin_amdgcn_global_load_async_to_lds_b128)
  __builtin_amdgcn_global_load_async_to_lds_b128(
      (__attribute__((address_space(1))) v4i*)(void*)const_cast<bf16_t*>(g),
      (__attribute__((address_space(3))) v4i*)(void*)l, 0, 0);
#else
  *(uint4*)l = *(const uint4*)g;     // load + ds_store staging fallback
#endif
}

__device__ __forceinline__ void wait_async_le1() {
#if __has_builtin(__builtin_amdgcn_s_wait_asynccnt)
  __builtin_amdgcn_s_wait_asynccnt(1);
#elif __has_builtin(__builtin_amdgcn_global_load_async_to_lds_b128)
  asm volatile("s_wait_asynccnt 0x1" ::: "memory");
#endif
}

__device__ __forceinline__ void wait_async_le0() {
#if __has_builtin(__builtin_amdgcn_s_wait_asynccnt)
  __builtin_amdgcn_s_wait_asynccnt(0);
#elif __has_builtin(__builtin_amdgcn_global_load_async_to_lds_b128)
  asm volatile("s_wait_asynccnt 0x0" ::: "memory");
#endif
}

// ---------------------------------------------------------------------------
// Convert fp32 W[K x N] (row-major) -> bf16 Wt[N x K] (row-major transpose).
// ---------------------------------------------------------------------------
__global__ __launch_bounds__(256) void k_conv_tr(const float* __restrict__ W,
                                                 bf16_t* __restrict__ Wt,
                                                 int K, int N) {
  long long gid = (long long)blockIdx.x * blockDim.x + threadIdx.x;
  long long total = (long long)K * N;
  if (gid >= total) return;
  int n = (int)(gid / K);
  int k = (int)(gid % K);
  Wt[gid] = f2bf(W[(size_t)k * N + n]);
}

// ---------------------------------------------------------------------------
// bf16 WMMA GEMM: C[M x N] = A[M x K] * Bt[N x K]^T + bias[N]  (fp32 acc)
// Block = 256 threads = 8 waves; block tile 128(M) x 64(N).
// Wave w: rows [16w,16w+16), 4 accumulators across the 64 N columns.
// B tile (64x32 bf16, 4KB) is async-DMA'd to LDS and double-buffered;
// A fragments come straight from global (distinct rows per wave, L2-hot).
// All 4 B fragments are loaded before the 4 WMMAs so the ds_loads clause up
// and the WMMAs issue back-to-back into the matrix pipe (single dscnt wait).
// grid = (N/64, M/128). Fragment layouts per CDNA5 ISA 7.12.2 (wave32).
// ---------------------------------------------------------------------------
__global__ __launch_bounds__(256) void k_gemm_bf16(const bf16_t* __restrict__ A,
                                                   const bf16_t* __restrict__ Bt,
                                                   const float*  __restrict__ bias,
                                                   float* __restrict__ C,
                                                   int M, int N, int K) {
  __shared__ __align__(16) bf16_t Bs[2][BN][ROWE];

  const int tid   = threadIdx.x;
  const int lane  = tid & 31;
  const int wave  = tid >> 5;
  const int nbase = blockIdx.x * BN;
  const int mbase = blockIdx.y * 128 + wave * 16;
  const int r  = lane & 15;   // row (A) / col (B) within 16-wide tile
  const int kh = lane >> 4;   // K-half select

  // B-tile copy mapping: thread -> (row n of 64, 16B segment of 64B row)
  const int cn  = tid >> 2;         // 0..63
  const int cs  = (tid & 3) * 8;    // element offset 0/8/16/24
  const bf16_t* gB = Bt + (size_t)(nbase + cn) * K + cs;

  // A lane holds K = kb..kb+7 and kb+16..kb+23 where kb = kh*8 (ISA layout)
  const bf16_t* arow = A + (size_t)(mbase + r) * K + kh * 8;

  const int nchunks = K / KC;

  // prologue: stage chunk 0 into buffer 0
  async_copy16(gB, &Bs[0][cn][cs]);

  v8f acc[4] = {v8f{}, v8f{}, v8f{}, v8f{}};

  for (int ch = 0; ch < nchunks; ++ch) {
    const int cur = ch & 1;
    if (ch + 1 < nchunks) {
      async_copy16(gB + (size_t)(ch + 1) * KC, &Bs[cur ^ 1][cn][cs]);
      wait_async_le1();
    } else {
      wait_async_le0();
    }
    __syncthreads();   // chunk ch resident in Bs[cur] for all waves

    const int kc = ch * KC;
    v8bf alo = *(const v8bf*)(arow + kc);
    v8bf ahi = *(const v8bf*)(arow + kc + 16);
    v16bf a;
#pragma unroll
    for (int i = 0; i < 8; ++i) { a[i] = alo[i]; a[i + 8] = ahi[i]; }

    // Load ALL B fragments first (8 clause-able ds_load_b128)...
    v16bf bfr[4];
#pragma unroll
    for (int s = 0; s < 4; ++s) {
      const bf16_t* bp = &Bs[cur][s * 16 + r][kh * 8];
      v8bf bl = *(const v8bf*)bp;
      v8bf bh = *(const v8bf*)(bp + 16);
#pragma unroll
      for (int i = 0; i < 8; ++i) { bfr[s][i] = bl[i]; bfr[s][i + 8] = bh[i]; }
    }
    // ...then fire the 4 WMMAs back-to-back.
#pragma unroll
    for (int s = 0; s < 4; ++s) {
      acc[s] = __builtin_amdgcn_wmma_f32_16x16x32_bf16(false, a, false, bfr[s],
                                                       (short)0, acc[s],
                                                       false, false);
    }
    __syncthreads();   // all waves done with Bs[cur] before it is re-filled
  }

  // C/D layout: lane -> col = lane&15; VGPR v -> row = v + (lane>>4)*8
  const int crow = mbase + kh * 8;
#pragma unroll
  for (int s = 0; s < 4; ++s) {
    const int ccol = nbase + s * 16 + r;
    const float bia = bias[ccol];
#pragma unroll
    for (int v = 0; v < 8; ++v) {
      C[(size_t)(crow + v) * N + ccol] = acc[s][v] + bia;
    }
  }
}

// ---------------------------------------------------------------------------
// Copy+convert current timestep input into concat buffer xin1[:, 0:128].
// ---------------------------------------------------------------------------
__global__ __launch_bounds__(256) void k_load_x(const float* __restrict__ inSeq,
                                                bf16_t* __restrict__ xin1, int t) {
  int idx = blockIdx.x * 256 + threadIdx.x;       // 0..16383
  int b = idx >> 7;
  int d = idx & 127;
  xin1[(size_t)b * (DIM + HID) + d] = f2bf(inSeq[(size_t)t * BATCH * DIM + idx]);
}

// ---------------------------------------------------------------------------
// LSTM cell pointwise: z = [i | j | f | o] (B x 4H fp32).
// ---------------------------------------------------------------------------
__global__ __launch_bounds__(256) void k_cell(const float* __restrict__ z,
                                              float* __restrict__ c,
                                              bf16_t* __restrict__ hA, int strideA,
                                              bf16_t* __restrict__ hB, int strideB) {
  int idx = blockIdx.x * 256 + threadIdx.x;       // 0..131071
  int b = idx >> 10;
  int j = idx & 1023;
  const float* zr = z + (size_t)b * 4096;
  float gi = zr[j];
  float gj = zr[1024 + j];
  float gf = zr[2048 + j];
  float go = zr[3072 + j];
  float cn = c[idx] * sigm(gf + 1.0f) + sigm(gi) * tanhf(gj);
  float h  = tanhf(cn) * sigm(go);
  c[idx] = cn;
  bf16_t hb = f2bf(h);
  hA[(size_t)b * strideA + j] = hb;
  hB[(size_t)b * strideB + j] = hb;
}

// ---------------------------------------------------------------------------
// In-place dual softmax over logits row (48 mel | 80 har) + per-row CE loss.
// ---------------------------------------------------------------------------
__global__ __launch_bounds__(128) void k_softmax_ce(float* __restrict__ io,
                                                    const int* __restrict__ tgt,
                                                    float* __restrict__ rowloss) {
  __shared__ float xs[128];
  __shared__ float es[128];
  __shared__ float red[4];   // melmax, harmax, melsum, harsum
  int row = blockIdx.x;
  int tid = threadIdx.x;
  float x = io[(size_t)row * 128 + tid];
  xs[tid] = x;
  __syncthreads();
  if (tid == 0)  { float m = xs[0];   for (int i = 1; i < MEL; ++i)       m = fmaxf(m, xs[i]); red[0] = m; }
  if (tid == 64) { float m = xs[MEL]; for (int i = MEL + 1; i < 128; ++i) m = fmaxf(m, xs[i]); red[1] = m; }
  __syncthreads();
  float gmax = (tid < MEL) ? red[0] : red[1];
  float e = __expf(x - gmax);
  es[tid] = e;
  __syncthreads();
  if (tid == 0)  { float s = 0.f; for (int i = 0; i < MEL; ++i)   s += es[i]; red[2] = s; }
  if (tid == 64) { float s = 0.f; for (int i = MEL; i < 128; ++i) s += es[i]; red[3] = s; }
  __syncthreads();
  float sum = (tid < MEL) ? red[2] : red[3];
  io[(size_t)row * 128 + tid] = e / sum;
  if (tid == 0) {
    int t0 = tgt[row * 2];
    int t1 = tgt[row * 2 + 1];
    float mel_lp = xs[t0]       - red[0] - __logf(red[2]);
    float har_lp = xs[MEL + t1] - red[1] - __logf(red[3]);
    rowloss[row] = 0.5f * (-mel_lp) + 0.5f * (-har_lp);
  }
}

// ---------------------------------------------------------------------------
// Deterministic loss reduction: fixed-order strided partials + LDS tree.
// ---------------------------------------------------------------------------
__global__ __launch_bounds__(256) void k_loss_reduce(const float* __restrict__ rowloss,
                                                     float* __restrict__ out) {
  __shared__ float s[256];
  float a = 0.f;
  for (int i = threadIdx.x; i < T_STEPS * BATCH; i += 256) a += rowloss[i];
  s[threadIdx.x] = a;
  __syncthreads();
  for (int st = 128; st > 0; st >>= 1) {
    if ((int)threadIdx.x < st) s[threadIdx.x] += s[threadIdx.x + st];
    __syncthreads();
  }
  if (threadIdx.x == 0) out[0] = s[0] / (float)(T_STEPS * BATCH);
}

// ---------------------------------------------------------------------------
extern "C" void kernel_launch(void* const* d_in, const int* in_sizes, int n_in,
                              void* d_out, int out_size, void* d_ws, size_t ws_size,
                              hipStream_t stream) {
  const float* inSeq   = (const float*)d_in[0];
  const int*   targets = (const int*)  d_in[1];
  const float* W1      = (const float*)d_in[2];
  const float* b1      = (const float*)d_in[3];
  const float* W2      = (const float*)d_in[4];
  const float* b2      = (const float*)d_in[5];
  const float* outW    = (const float*)d_in[6];
  const float* outB    = (const float*)d_in[7];
  float* out = (float*)d_out;

  char* ws = (char*)d_ws;
  size_t off = 0;
  auto alloc = [&](size_t bytes) -> void* {
    void* p = ws + off;
    off += (bytes + 255) & ~(size_t)255;
    return p;
  };

  bf16_t* W1t   = (bf16_t*)alloc((size_t)4096 * 1152 * 2);   // [4H x (D+H)]
  bf16_t* W2t   = (bf16_t*)alloc((size_t)4096 * 2048 * 2);   // [4H x 2H]
  bf16_t* outWt = (bf16_t*)alloc((size_t)128  * 1024 * 2);   // [D x H]
  bf16_t* xin1  = (bf16_t*)alloc((size_t)BATCH * (DIM + HID) * 2);  // [x | h1]
  bf16_t* xin2  = (bf16_t*)alloc((size_t)BATCH * (2 * HID) * 2);    // [h1 | h2]
  float*  c1    = (float*) alloc((size_t)BATCH * HID * 4);
  float*  c2    = (float*) alloc((size_t)BATCH * HID * 4);
  float*  z1    = (float*) alloc((size_t)BATCH * 4 * HID * 4);
  float*  z2    = (float*) alloc((size_t)BATCH * 4 * HID * 4);
  bf16_t* h2all = (bf16_t*)alloc((size_t)T_STEPS * BATCH * HID * 2);
  float*  rowls = (float*) alloc((size_t)T_STEPS * BATCH * 4);

  // One-time per call: bf16 transposed weights (L2-resident afterwards).
  k_conv_tr<<<(4096 * 1152 + 255) / 256, 256, 0, stream>>>(W1, W1t, 1152, 4096);
  k_conv_tr<<<(4096 * 2048 + 255) / 256, 256, 0, stream>>>(W2, W2t, 2048, 4096);
  k_conv_tr<<<(128 * 1024 + 255) / 256, 256, 0, stream>>>(outW, outWt, 1024, 128);

  // Zero initial state: xin1, xin2, c1, c2 are contiguous in the workspace.
  size_t zeroBytes = (size_t)BATCH * (DIM + HID) * 2 + (size_t)BATCH * 2 * HID * 2
                   + (size_t)BATCH * HID * 4 * 2;
  (void)hipMemsetAsync(xin1, 0, zeroBytes, stream);

  dim3 gRec(4096 / BN, 1, 1);       // recurrent GEMMs: 64 blocks
  for (int t = 0; t < T_STEPS; ++t) {
    k_load_x<<<(BATCH * DIM) / 256, 256, 0, stream>>>(inSeq, xin1, t);
    // layer 1: z1 = [x|h1] @ W1 + b1
    k_gemm_bf16<<<gRec, 256, 0, stream>>>(xin1, W1t, b1, z1, BATCH, 4096, DIM + HID);
    k_cell<<<(BATCH * HID) / 256, 256, 0, stream>>>(z1, c1,
        xin1 + DIM, DIM + HID,          // h1 -> xin1[:,128:1152] (next step)
        xin2, 2 * HID);                 // h1 -> xin2[:,0:1024]
    // layer 2: z2 = [h1|h2] @ W2 + b2
    k_gemm_bf16<<<gRec, 256, 0, stream>>>(xin2, W2t, b2, z2, BATCH, 4096, 2 * HID);
    k_cell<<<(BATCH * HID) / 256, 256, 0, stream>>>(z2, c2,
        xin2 + HID, 2 * HID,            // h2 -> xin2[:,1024:2048] (next step)
        h2all + (size_t)t * BATCH * HID, HID);
  }

  // Final projection: logits straight into d_out, then in-place softmax.
  dim3 gOut(128 / BN, (T_STEPS * BATCH) / 128, 1);   // (2, 256)
  k_gemm_bf16<<<gOut, 256, 0, stream>>>(h2all, outWt, outB, out,
                                        T_STEPS * BATCH, 128, HID);
  k_softmax_ce<<<T_STEPS * BATCH, 128, 0, stream>>>(out, targets, rowls);
  k_loss_reduce<<<1, 256, 0, stream>>>(rowls, out + (size_t)T_STEPS * BATCH * DIM);
}
